// RNN_10067403342387
// MI455X (gfx1250) — matrix operand order repaired and checked by reading, not compiled, not run on hardware
//
#include <hip/hip_runtime.h>
#include <hip/hip_bf16.h>

// ---------------------------------------------------------------------------
// Types for CDNA5 WMMA (wave32): 16x16x32 bf16 -> f32 accumulate
// ---------------------------------------------------------------------------
typedef __attribute__((ext_vector_type(16))) __bf16        v16bf;
typedef __attribute__((ext_vector_type(8)))  float         v8f;
typedef __attribute__((ext_vector_type(8)))  unsigned int  v8u;   // 32B raw

// explicit global (addrspace 1) views to force global_load_b128 (not flat)
typedef __attribute__((address_space(1))) const unsigned short gus_t;
typedef __attribute__((address_space(1))) const v8u            gv8u_t;

#define B_    256
#define T_    1024
#define D_    128
#define H_    512
#define G_    2048          // 4*H
#define C_    10
#define KTOT  640           // D + H
#define NKF   20            // KTOT / 32
#define NBT   16            // batch tiles of M=16
#define NNT   128           // 4H / 16 column tiles

// workspace layout (bf16 stored as unsigned short)
#define WSW_SHORTS   ((size_t)NNT * NKF * 512)        //  2.62 MB: weight frags
#define WSX_SHORTS   ((size_t)NBT * T_ * 4 * 512)     // 64   MB: x frags
#define WSX_BYTE_OFF (WSW_SHORTS * 2)
#define HFIN_BYTE_OFF (WSX_BYTE_OFF + WSX_SHORTS * 2) // 512 KB final h (fp32)

// ---------------------------------------------------------------------------
// helpers
// ---------------------------------------------------------------------------
__device__ __forceinline__ unsigned short f2bf(float f) {
  unsigned u = __builtin_bit_cast(unsigned, f);
  unsigned r = (u + 0x7FFFu + ((u >> 16) & 1u)) >> 16;  // RNE
  return (unsigned short)r;
}

// ISA 7.12.2 16-bit A-matrix 16x32 layout: lane holds row m = lane&15;
// element i (i-th bf16 in the 8 VGPRs) maps to k_local:
__device__ __forceinline__ int klocal_A(int lane, int i) {
  return i + (i & 8) + ((lane & 16) >> 1);
}

__device__ __forceinline__ float sigmoid_f(float x) {
  return 1.0f / (1.0f + __expf(-x));
}
__device__ __forceinline__ float tanh_f(float x) {
  float e = __expf(-2.0f * fabsf(x));
  float t = (1.0f - e) / (1.0f + e);
  return __builtin_copysignf(t, x);
}

// ---------------------------------------------------------------------------
// Prep 1: swizzle [W_ih | W_hh] (G x KTOT) into B-fragment order, bf16.
// ---------------------------------------------------------------------------
__global__ void swizzle_weights(const float* __restrict__ Wih,
                                const float* __restrict__ Whh,
                                unsigned short* __restrict__ wsw) {
  int idx  = blockIdx.x * 256 + threadIdx.x;        // < NNT*NKF*512
  int e    = idx & 511;
  int kf   = (idx >> 9) % NKF;
  int nt   = idx / (512 * NKF);
  int lane = e >> 4, i = e & 15;
  int g    = nt * 16 + (lane & 15);
  int k    = kf * 32 + klocal_A(lane, i);
  float v  = (k < D_) ? Wih[(size_t)g * D_ + k] : Whh[(size_t)g * H_ + (k - D_)];
  wsw[idx] = f2bf(v);
}

// ---------------------------------------------------------------------------
// Prep 2: swizzle x [B,T,D] into A-fragment order, bf16.
// ---------------------------------------------------------------------------
__global__ void swizzle_x(const float* __restrict__ x,
                          unsigned short* __restrict__ wsx) {
  size_t idx = (size_t)blockIdx.x * 256 + threadIdx.x;  // < NBT*T*4*512
  int e    = idx & 511;
  int kf   = (int)((idx >> 9) & 3);
  int t    = (int)((idx >> 11) & 1023);
  int bt   = (int)(idx >> 21);
  int lane = e >> 4, i = e & 15;
  int m    = lane & 15;
  int k    = kf * 32 + klocal_A(lane, i);
  float v  = x[((size_t)(bt * 16 + m) * T_ + t) * D_ + k];
  wsx[idx] = f2bf(v);
}

// ---------------------------------------------------------------------------
// Persistent LSTM scan: one WG per batch tile of 16 rows, 16 waves (512 thr).
// Wave w owns hidden slices s = 2w, 2w+1 (16 cols each) across all 4 gates.
// h fragments double-buffered in LDS; c state stays in registers.
// Weights stream from L2 every step (2.5 MB/WG/step); an empty-asm barrier on
// a scalar OFFSET (not the pointer!) keeps LICM from hoisting/spilling the
// weight fragments while preserving global address-space inference.
// ---------------------------------------------------------------------------
__global__ void __launch_bounds__(512, 1)
lstm_scan(const unsigned short* __restrict__ wsx,
          const unsigned short* __restrict__ wsw,
          const float* __restrict__ bias_g,
          float* __restrict__ hfin) {
  __shared__ alignas(32) unsigned short hfrag[2][16][512];  // 32 KB

  const int tid  = threadIdx.x;
  const int lane = tid & 31;
  const int wave = tid >> 5;
  const int bt   = blockIdx.x;
  const int n    = lane & 15;

  // zero-init h(0) fragments
  {
    unsigned* z = (unsigned*)&hfrag[0][0][0];
    for (int j = tid; j < 16 * 512 / 2; j += 512) z[j] = 0u;
  }
  __syncthreads();

  // per-tile bias (column-uniform across the 8 acc VGPRs)
  float bias[4][2];
#pragma unroll
  for (int q = 0; q < 4; ++q)
#pragma unroll
    for (int sl = 0; sl < 2; ++sl)
      bias[q][sl] = bias_g[q * H_ + (wave * 2 + sl) * 16 + n];

  v8f c[2];
#pragma unroll
  for (int sl = 0; sl < 2; ++sl)
#pragma unroll
    for (int r = 0; r < 8; ++r) c[sl][r] = 0.0f;

  const unsigned short* xbase = wsx + (size_t)bt * T_ * 4 * 512 + lane * 16;
  int par = 0;

  for (int t = 0; t < T_; ++t) {
    // Opaque scalar offset (always 0): makes weight addresses formally
    // loop-variant (no LICM hoist/spill) while the pointer chain stays rooted
    // at the kernel arg, so loads remain addrspace(1) -> global_load_b128.
    int wofs = 0;
    asm volatile("" : "+s"(wofs));
    const gus_t* wlane = (const gus_t*)(wsw + wofs) + lane * 16;

    // prefetch next timestep's input fragments toward near caches
    if (t + 1 < T_) {
#pragma unroll
      for (int kf = 0; kf < 4; ++kf)
        __builtin_prefetch(xbase + ((size_t)(t + 1) * 4 + kf) * 512, 0, 3);
    }

    v8f acc[4][2];
#pragma unroll
    for (int q = 0; q < 4; ++q)
#pragma unroll
      for (int sl = 0; sl < 2; ++sl)
#pragma unroll
        for (int r = 0; r < 8; ++r) acc[q][sl][r] = bias[q][sl];

    // ---- input projection: K = 0..127 from pre-swizzled x fragments ----
#pragma unroll
    for (int kf = 0; kf < 4; ++kf) {
      v16bf A = __builtin_bit_cast(
          v16bf, *(const v8u*)(xbase + ((size_t)t * 4 + kf) * 512));
      v8u Braw[8];
#pragma unroll
      for (int j = 0; j < 8; ++j) {           // j = q*2 + sl
        int ntile = (j >> 1) * 32 + wave * 2 + (j & 1);
        Braw[j] = *(const gv8u_t*)(wlane + ((size_t)ntile * NKF + kf) * 512);
      }
#pragma unroll
      for (int j = 0; j < 8; ++j) {
        v16bf Bf = __builtin_bit_cast(v16bf, Braw[j]);
        acc[j >> 1][j & 1] = __builtin_amdgcn_wmma_f32_16x16x32_bf16(
            false, A, false, Bf, (short)0, acc[j >> 1][j & 1], false, false);
      }
    }

    // ---- recurrence: K = 128..639 from LDS h fragments ----
#pragma unroll 2
    for (int kf = 4; kf < NKF; ++kf) {
      v16bf A = __builtin_bit_cast(
          v16bf, *(const v8u*)&hfrag[par][kf - 4][lane * 16]);
      v8u Braw[8];
#pragma unroll
      for (int j = 0; j < 8; ++j) {
        int ntile = (j >> 1) * 32 + wave * 2 + (j & 1);
        Braw[j] = *(const gv8u_t*)(wlane + ((size_t)ntile * NKF + kf) * 512);
      }
#pragma unroll
      for (int j = 0; j < 8; ++j) {
        v16bf Bf = __builtin_bit_cast(v16bf, Braw[j]);
        acc[j >> 1][j & 1] = __builtin_amdgcn_wmma_f32_16x16x32_bf16(
            false, A, false, Bf, (short)0, acc[j >> 1][j & 1], false, false);
      }
    }

    // ---- LSTM cell update (gate order i,f,g,o); write h(t+1) swizzled ----
#pragma unroll
    for (int sl = 0; sl < 2; ++sl) {
      const int s  = wave * 2 + sl;
      const int kl = ((s & 1) << 4) | n;  // k_local of this hidden column
#pragma unroll
      for (int r = 0; r < 8; ++r) {
        float iv = sigmoid_f(acc[0][sl][r]);
        float fv = sigmoid_f(acc[1][sl][r]);
        float gv = tanh_f(acc[2][sl][r]);
        float ov = sigmoid_f(acc[3][sl][r]);
        float cv = fv * c[sl][r] + iv * gv;
        c[sl][r] = cv;
        float hv = ov * tanh_f(cv);

        int m = (lane < 16) ? r : r + 8;  // C/D layout row
        int a_lane, ii;                   // inverse of klocal_A
        if (kl < 8)       { a_lane = m;      ii = kl;      }
        else if (kl < 16) { a_lane = m + 16; ii = kl - 8;  }
        else if (kl < 24) { a_lane = m;      ii = kl - 8;  }
        else              { a_lane = m + 16; ii = kl - 16; }
        // this wave's hidden slice maps exactly to k-fragment row `wave`
        hfrag[par ^ 1][wave][a_lane * 16 + ii] = f2bf(hv);

        if (t == T_ - 1)
          hfin[(size_t)(bt * 16 + m) * H_ + s * 16 + n] = hv;
      }
    }
    __syncthreads();
    par ^= 1;
  }
}

// ---------------------------------------------------------------------------
// Classifier: logits = h @ W_out^T + b_out, softmax. Tiny (256x10x512).
// ---------------------------------------------------------------------------
__global__ void __launch_bounds__(256)
classifier(const float* __restrict__ hfin, const float* __restrict__ Wout,
           const float* __restrict__ bout, float* __restrict__ out) {
  __shared__ float w[C_][H_];
  for (int j = threadIdx.x; j < C_ * H_; j += 256) ((float*)w)[j] = Wout[j];
  __syncthreads();

  int row = threadIdx.x;  // grid = 1 block of 256
  const float* h = hfin + (size_t)row * H_;
  float logit[C_];
#pragma unroll
  for (int cc = 0; cc < C_; ++cc) {
    float s = bout[cc];
    for (int k = 0; k < H_; ++k) s = fmaf(h[k], w[cc][k], s);
    logit[cc] = s;
  }
  float mx = logit[0];
#pragma unroll
  for (int cc = 1; cc < C_; ++cc) mx = fmaxf(mx, logit[cc]);
  float sum = 0.0f;
#pragma unroll
  for (int cc = 0; cc < C_; ++cc) {
    logit[cc] = __expf(logit[cc] - mx);
    sum += logit[cc];
  }
  float inv = 1.0f / sum;
#pragma unroll
  for (int cc = 0; cc < C_; ++cc) out[(size_t)row * C_ + cc] = logit[cc] * inv;
}

// ---------------------------------------------------------------------------
extern "C" void kernel_launch(void* const* d_in, const int* in_sizes, int n_in,
                              void* d_out, int out_size, void* d_ws,
                              size_t ws_size, hipStream_t stream) {
  (void)in_sizes; (void)n_in; (void)out_size; (void)ws_size;
  const float* x     = (const float*)d_in[0];
  const float* W_ih  = (const float*)d_in[1];
  const float* W_hh  = (const float*)d_in[2];
  const float* b     = (const float*)d_in[3];
  const float* W_out = (const float*)d_in[4];
  const float* b_out = (const float*)d_in[5];

  unsigned short* wsw  = (unsigned short*)d_ws;
  unsigned short* wsx  = (unsigned short*)((char*)d_ws + WSX_BYTE_OFF);
  float*          hfin = (float*)((char*)d_ws + HFIN_BYTE_OFF);

  swizzle_weights<<<(int)(WSW_SHORTS / 256), 256, 0, stream>>>(W_ih, W_hh, wsw);
  swizzle_x<<<(int)(WSX_SHORTS / 256), 256, 0, stream>>>(x, wsx);
  lstm_scan<<<NBT, 512, 0, stream>>>(wsx, wsw, b, hfin);
  classifier<<<1, 256, 0, stream>>>(hfin, W_out, b_out, (float*)d_out);
}
